// ODE_RNN_4011499455056
// MI455X (gfx1250) — matrix operand order repaired and checked by reading, not compiled, not run on hardware
//
#include <hip/hip_runtime.h>
#include <hip/hip_bf16.h>
#include <stdint.h>

// Problem dims (fixed by reference)
#define SEQ   512
#define IND   1024
#define HID   2048
#define OUTD  128
#define NSUB  20

// Kernel config
#define NWG       32     // persistent workgroups; each owns HID/NWG = 64 rows
#define ROWS      64     // hidden rows per WG
#define NTHREADS  256    // 8 wave32 waves: 4 row-tiles x 2 K-halves
#define WSTRIDE   4112   // LDS bytes per W_ode row (2048 bf16 = 4096B + 16B pad vs bank conflicts)

typedef __attribute__((ext_vector_type(16))) __bf16 v16bf;
typedef __attribute__((ext_vector_type(8)))  float  v8f;

// ---- LDS layout (dynamic shared memory), all 16B aligned ----
#define OFF_WODE 0                           // 64 rows x 4112B  = 263168
#define OFF_HBF  (ROWS*WSTRIDE)              // bf16 stage-input vector, 4096B
#define OFF_XBF  (OFF_HBF + 4096)            // bf16 x_t vector, 2048B
#define OFF_YRED (OFF_XBF + 2048)            // 8 waves x 16 floats = 512B
#define OFF_KARR (OFF_YRED + 512)            // k1..k6 slices: 6*64*4 = 1536B
#define OFF_HARR (OFF_KARR + 6*ROWS*4)       // h slice fp32, 256B
#define OFF_BODE (OFF_HARR + ROWS*4)         // b_ode slice, 256B
#define OFF_BSUM (OFF_BODE + ROWS*4)         // b_in+b_h slice, 256B
#define LDS_BYTES (OFF_BSUM + ROWS*4)        // = 272128 B (~266 KB) < 320 KB

// dopri5 tableau
__device__ __constant__ float d_AC[6][5] = {
  {0.f,0.f,0.f,0.f,0.f},
  {1.f/5.f,0.f,0.f,0.f,0.f},
  {3.f/40.f, 9.f/40.f,0.f,0.f,0.f},
  {44.f/45.f, -56.f/15.f, 32.f/9.f,0.f,0.f},
  {19372.f/6561.f, -25360.f/2187.f, 64448.f/6561.f, -212.f/729.f,0.f},
  {9017.f/3168.f, -355.f/33.f, 46732.f/5247.f, 49.f/176.f, -5103.f/18656.f}
};
__device__ __constant__ float d_BC[6] = {
  35.f/384.f, 0.f, 500.f/1113.f, 125.f/192.f, -2187.f/6784.f, 11.f/84.f
};

// round-to-nearest-even f32 -> bf16 helpers
__device__ __forceinline__ uint32_t pk_bf16(float a, float b){
  uint32_t ua = __float_as_uint(a), ub = __float_as_uint(b);
  ua += 0x7FFFu + ((ua >> 16) & 1u);
  ub += 0x7FFFu + ((ub >> 16) & 1u);
  return (ua >> 16) | (ub & 0xFFFF0000u);
}
__device__ __forceinline__ uint16_t bf16of(float a){
  uint32_t ua = __float_as_uint(a);
  ua += 0x7FFFu + ((ua >> 16) & 1u);
  return (uint16_t)(ua >> 16);
}

__device__ __forceinline__ v8f wmma_bf16(uint4 a0, uint4 a1, uint4 b0, uint4 b1, v8f c){
  union { uint4 q[2]; v16bf v; } A, B;
  A.q[0] = a0; A.q[1] = a1; B.q[0] = b0; B.q[1] = b1;
  return __builtin_amdgcn_wmma_f32_16x16x32_bf16(false, A.v, false, B.v, (short)0, c, false, false);
}

// GEMV K-block accumulate, weights resident in LDS as bf16 (used 120x per timestep).
// Software-pipelined: next block's 4x ds_load_b128 issue before this block's WMMA,
// so the MMA waits on dscnt<=4 instead of dscnt==0.
__device__ __forceinline__ v8f lds_acc(const char* wode, const char* hbf,
                                       int rowtile, int kb0, int kbn, v8f acc){
  const int l  = threadIdx.x & 31;
  const int n  = l & 15;
  const int hi = (l >> 4) & 1;
  const char* brow = wode + (rowtile*16 + n)*WSTRIDE + hi*32;
  const char* ab   = hbf + hi*32;

  uint4 a0 = *(const uint4*)(ab + kb0*64);
  uint4 a1 = *(const uint4*)(ab + kb0*64 + 16);
  uint4 b0 = *(const uint4*)(brow + kb0*64);
  uint4 b1 = *(const uint4*)(brow + kb0*64 + 16);
  #pragma unroll 4
  for (int kb = kb0; kb < kb0 + kbn - 1; ++kb){
    uint4 na0 = *(const uint4*)(ab + (kb+1)*64);
    uint4 na1 = *(const uint4*)(ab + (kb+1)*64 + 16);
    uint4 nb0 = *(const uint4*)(brow + (kb+1)*64);
    uint4 nb1 = *(const uint4*)(brow + (kb+1)*64 + 16);
    acc = wmma_bf16(a0, a1, b0, b1, acc);
    a0 = na0; a1 = na1; b0 = nb0; b1 = nb1;
  }
  return wmma_bf16(a0, a1, b0, b1, acc);
}

// GEMV K-block accumulate, fp32 weights streamed from global (L2) with inline bf16 pack.
__device__ __forceinline__ v8f stream_acc(const float* __restrict__ W, int K,
                                          const char* aLDS, int rowbase,
                                          int kb0, int kbn, v8f acc){
  const int l  = threadIdx.x & 31;
  const int n  = l & 15;
  const int hi = (l >> 4) & 1;
  const float* wrow = W + (size_t)(rowbase + n) * (size_t)K + hi*16;
  const char*  ab   = aLDS + hi*32;
  for (int kb = kb0; kb < kb0 + kbn; ++kb){
    uint4 a0 = *(const uint4*)(ab + kb*64);
    uint4 a1 = *(const uint4*)(ab + kb*64 + 16);
    const float* wp = wrow + kb*32;
    float4 f0 = *(const float4*)(wp +  0);
    float4 f1 = *(const float4*)(wp +  4);
    float4 f2 = *(const float4*)(wp +  8);
    float4 f3 = *(const float4*)(wp + 12);
    uint4 b0, b1;
    b0.x = pk_bf16(f0.x, f0.y); b0.y = pk_bf16(f0.z, f0.w);
    b0.z = pk_bf16(f1.x, f1.y); b0.w = pk_bf16(f1.z, f1.w);
    b1.x = pk_bf16(f2.x, f2.y); b1.y = pk_bf16(f2.z, f2.w);
    b1.z = pk_bf16(f3.x, f3.y); b1.w = pk_bf16(f3.z, f3.w);
    acc = wmma_bf16(a0, a1, b0, b1, acc);
  }
  return acc;
}

// Combine the 2 K-half partials per row-tile; returns y for tid<ROWS (row = tid).
// No leading sync needed: consecutive reductions are always separated by the
// grid barrier (two workgroup syncs), so yred reuse cannot race.
__device__ __forceinline__ float reduce16(v8f acc, float* yred){
  const int tid = threadIdx.x;
  const int wave = tid >> 5, l = tid & 31;
  if (l < 16) yred[wave*16 + l] = acc[0];   // D VGPR0, lanes 0..15 = (M=0, N=lane)
  __syncthreads();
  float y = 0.f;
  if (tid < ROWS){
    const int rt = tid >> 4, n = tid & 15;
    y = yred[(rt*2 + 0)*16 + n] + yred[(rt*2 + 1)*16 + n];
  }
  return y;
}

// monotonic-counter grid barrier (all NWG workgroups resident: 1 WG/WGP)
__device__ __forceinline__ void grid_barrier(unsigned* ctr, unsigned target){
  __syncthreads();
  if (threadIdx.x == 0){
    __hip_atomic_fetch_add(ctr, 1u, __ATOMIC_RELEASE, __HIP_MEMORY_SCOPE_AGENT);
    while (__hip_atomic_load(ctr, __ATOMIC_ACQUIRE, __HIP_MEMORY_SCOPE_AGENT) < target)
      __builtin_amdgcn_s_sleep(1);
  }
  __syncthreads();
}

__global__ void init_ctr(unsigned* ctr){
  if (threadIdx.x == 0 && blockIdx.x == 0) *ctr = 0u;
}

__global__ void __launch_bounds__(NTHREADS, 1)
ode_rnn_persistent(const float* __restrict__ tg,   const float* __restrict__ xg,
                   const float* __restrict__ Win,  const float* __restrict__ bin,
                   const float* __restrict__ Wh,   const float* __restrict__ bh,
                   const float* __restrict__ Wode, const float* __restrict__ bode_g,
                   const float* __restrict__ Wdec, const float* __restrict__ bdec,
                   float* __restrict__ out, char* __restrict__ ws)
{
  extern __shared__ char smem[];
  char*  wode = smem + OFF_WODE;
  char*  hbf  = smem + OFF_HBF;
  char*  xbf  = smem + OFF_XBF;
  float* yred = (float*)(smem + OFF_YRED);
  float* karr = (float*)(smem + OFF_KARR);   // [6][ROWS]
  float* harr = (float*)(smem + OFF_HARR);
  float* bode = (float*)(smem + OFF_BODE);
  float* bsum = (float*)(smem + OFF_BSUM);

  unsigned* ctr  = (unsigned*)ws;
  uint16_t* hbuf = (uint16_t*)(ws + 256);           // double buffer [2][HID] bf16 (8 KB)
  float*    hfin = (float*)(ws + 256 + 2*HID*2);    // final h fp32 (8 KB)

  const int tid     = threadIdx.x;
  const int wg      = blockIdx.x;
  const int rowbase = wg * ROWS;
  const int wave    = tid >> 5;
  const int rowtile = wave >> 1;     // 0..3 -> rows rowtile*16 .. +15
  const int khalf   = wave & 1;      // K split across wave pairs

  // ---- setup: W_ode slice (64 rows) fp32 -> bf16 resident in LDS ----
  for (int r = 0; r < ROWS; ++r){
    const float* src = Wode + (size_t)(rowbase + r)*HID + tid*8;
    float4 f0 = *(const float4*)(src);
    float4 f1 = *(const float4*)(src + 4);
    uint4 v;
    v.x = pk_bf16(f0.x, f0.y); v.y = pk_bf16(f0.z, f0.w);
    v.z = pk_bf16(f1.x, f1.y); v.w = pk_bf16(f1.z, f1.w);
    *(uint4*)(wode + r*WSTRIDE + tid*16) = v;
  }
  if (tid < ROWS){
    bode[tid] = bode_g[rowbase + tid];
    bsum[tid] = bin[rowbase + tid] + bh[rowbase + tid];
  }
  __syncthreads();

  unsigned phase = 0;
  int p = 0;

  // ---- initial hidden: h0 = tanh(x0 @ Win.T + b_in + b_h)  (local, no barrier) ----
  {
    const float* xp = xg + tid*4;
    float4 f = *(const float4*)xp;
    uint2 v; v.x = pk_bf16(f.x, f.y); v.y = pk_bf16(f.z, f.w);
    *(uint2*)(xbf + tid*8) = v;
  }
  __syncthreads();
  {
    v8f acc = {};
    acc = stream_acc(Win, IND, xbf, rowbase, khalf*16, 16, acc);
    float y = reduce16(acc, yred);
    if (tid < ROWS) harr[tid] = tanhf(y + bsum[tid]);
  }

  // ---- sequential scan over timesteps ----
  for (int i = 1; i < SEQ; ++i){
    const float hs = (tg[i] - tg[i-1]) * (1.0f / (float)NSUB);

    // warm the L2/WGP path for this timestep's streamed operands
    __builtin_prefetch(xg + (size_t)i*IND + tid*4, 0, 1);

    // 20 dopri5 substeps: 6 stages each, W_ode GEMV from LDS via WMMA
    for (int sub = 0; sub < NSUB; ++sub){
      // rolling prefetch window over this WG's W_h slice (used after the substeps)
      __builtin_prefetch((const char*)(Wh + (size_t)rowbase*HID) +
                         ((size_t)sub*NTHREADS + tid)*64, 0, 1);

      for (int s = 0; s < 6; ++s){
        if (tid < ROWS){
          float z = harr[tid];
          #pragma unroll
          for (int j = 0; j < 5; ++j)
            if (j < s) z += hs * d_AC[s][j] * karr[j*ROWS + tid];
          hbuf[p*HID + rowbase + tid] = bf16of(z);   // publish slice (bf16)
        }
        ++phase; grid_barrier(ctr, phase*NWG);       // all slices visible
        { // pull full 4KB stage vector into LDS
          uint4 v = *(const uint4*)((const char*)(hbuf + p*HID) + tid*16);
          *(uint4*)(hbf + tid*16) = v;
        }
        p ^= 1;
        __syncthreads();
        v8f acc = {};
        acc = lds_acc(wode, hbf, rowtile, khalf*32, 32, acc);  // 32 WMMAs/wave
        float y = reduce16(acc, yred);
        if (tid < ROWS) karr[s*ROWS + tid] = tanhf(y + bode[tid]);
      }
      if (tid < ROWS){
        float h = harr[tid];
        #pragma unroll
        for (int j = 0; j < 6; ++j) h += hs * d_BC[j] * karr[j*ROWS + tid];
        harr[tid] = h;
      }
    }

    // h_new = tanh(x_i @ Win.T + b_in + h_ode @ Wh.T + b_h): stream Wh/Win from L2
    if (tid < ROWS) hbuf[p*HID + rowbase + tid] = bf16of(harr[tid]);
    {
      const float* xp = xg + (size_t)i*IND + tid*4;
      float4 f = *(const float4*)xp;
      uint2 v; v.x = pk_bf16(f.x, f.y); v.y = pk_bf16(f.z, f.w);
      *(uint2*)(xbf + tid*8) = v;
    }
    ++phase; grid_barrier(ctr, phase*NWG);
    {
      uint4 v = *(const uint4*)((const char*)(hbuf + p*HID) + tid*16);
      *(uint4*)(hbf + tid*16) = v;
    }
    p ^= 1;
    __syncthreads();
    v8f acc = {};
    acc = stream_acc(Wh,  HID, hbf, rowbase, khalf*32, 32, acc);
    acc = stream_acc(Win, IND, xbf, rowbase, khalf*16, 16, acc);
    float y = reduce16(acc, yred);
    if (tid < ROWS) harr[tid] = tanhf(y + bsum[tid]);
  }

  // ---- decoder: out = h @ Wdec.T + b_dec ----
  if (tid < ROWS) hfin[rowbase + tid] = harr[tid];
  ++phase; grid_barrier(ctr, phase*NWG);
  if (wg == 0){
    const int o    = tid & 127;
    const int half = tid >> 7;
    const float* wd = Wdec + (size_t)o*HID + half*1024;
    const float* hp = hfin + half*1024;
    float s = 0.f;
    for (int k = 0; k < 1024; k += 4){
      float4 w4 = *(const float4*)(wd + k);
      float4 h4 = *(const float4*)(hp + k);
      s += w4.x*h4.x + w4.y*h4.y + w4.z*h4.z + w4.w*h4.w;
    }
    float* dec = karr;          // reuse LDS scratch (256 floats)
    dec[tid] = s;
    __syncthreads();
    if (tid < OUTD) out[tid] = dec[tid] + dec[tid + 128] + bdec[tid];
  }
}

extern "C" void kernel_launch(void* const* d_in, const int* in_sizes, int n_in,
                              void* d_out, int out_size, void* d_ws, size_t ws_size,
                              hipStream_t stream){
  (void)in_sizes; (void)n_in; (void)out_size; (void)ws_size;
  const float* tg   = (const float*)d_in[0];
  const float* xg   = (const float*)d_in[1];
  const float* Win  = (const float*)d_in[2];
  const float* bin  = (const float*)d_in[3];
  const float* Wh   = (const float*)d_in[4];
  const float* bh   = (const float*)d_in[5];
  const float* Wode = (const float*)d_in[6];
  const float* bode = (const float*)d_in[7];
  const float* Wdec = (const float*)d_in[8];
  const float* bdec = (const float*)d_in[9];

  init_ctr<<<1, 32, 0, stream>>>((unsigned*)d_ws);
  ode_rnn_persistent<<<NWG, NTHREADS, LDS_BYTES, stream>>>(
      tg, xg, Win, bin, Wh, bh, Wode, bode, Wdec, bdec,
      (float*)d_out, (char*)d_ws);
}
